// MultiHeadAttention_39771397161295
// MI455X (gfx1250) — matrix operand order repaired and checked by reading, not compiled, not run on hardware
//
#include <hip/hip_runtime.h>

// ---------------------------------------------------------------------------
// Types / WMMA helper (CDNA5 / gfx1250, wave32)
// ---------------------------------------------------------------------------
typedef __attribute__((ext_vector_type(16))) __bf16 bf16x16;
typedef __attribute__((ext_vector_type(8)))  __bf16 bf16x8;
typedef __attribute__((ext_vector_type(8)))  float  f32x8;

union Frag16 { bf16x16 v; bf16x8 h[2]; };

static __device__ __forceinline__ f32x8 wmma_bf16(bf16x16 a, bf16x16 b, f32x8 c) {
  // (neg_a, A, neg_b, B, c_mod, C, reuse_a, reuse_b)
  return __builtin_amdgcn_wmma_f32_16x16x32_bf16(false, a, false, b, (short)0, c,
                                                 false, false);
}

// ds_swizzle butterfly max-reduce across 16-lane groups, batched per step so
// the compiler can issue 8 swizzles under a single s_wait_dscnt.
template <int IMM>
static __device__ __forceinline__ void swz_max8(float mx[8]) {
  float tmp[8];
#pragma unroll
  for (int r = 0; r < 8; ++r)
    tmp[r] = __int_as_float(
        __builtin_amdgcn_ds_swizzle(__float_as_int(mx[r]), IMM));
#pragma unroll
  for (int r = 0; r < 8; ++r) mx[r] = fmaxf(mx[r], tmp[r]);
}

// Optional async global->LDS staging (CDNA5 ASYNCcnt path), guarded so the
// file compiles on toolchains without the builtin.
#if defined(__has_builtin)
#if __has_builtin(__builtin_amdgcn_global_load_async_to_lds_b128)
#define HAVE_ASYNC_LDS 1
#endif
#endif
#ifndef HAVE_ASYNC_LDS
#define HAVE_ASYNC_LDS 0
#endif

#if HAVE_ASYNC_LDS
// Signature learned from hipcc diagnostic:
//   (v4i addrspace(1)*, v4i addrspace(3)*, int offset, int cpol)
typedef int gld_v4i __attribute__((vector_size(16)));
typedef __attribute__((address_space(1))) gld_v4i* as1_v4i_p;
typedef __attribute__((address_space(3))) gld_v4i* as3_v4i_p;
#endif

#define D_MODEL 1024
#define SEQ     2048
#define NB      2
#define NH      16
#define DK      64

// ---------------------------------------------------------------------------
// Tiled GEMM: out[M x 1024] = alpha * (A[M x 1024] @ W[1024 x 1024]) + bias
// A: fp32 or bf16 (converted to bf16 in LDS); out: bf16 or fp32.
// Block: 256 threads (8 waves). Tile: BM=128, BN=64, BK=32.
// Each wave: 32x32 output = 2x2 WMMA(16x16) tiles, f32 accumulate.
// ---------------------------------------------------------------------------
template <typename AT, typename OT>
__global__ __launch_bounds__(256)
void gemm_bf16_wmma(const AT* __restrict__ A, const float* __restrict__ W,
                    const float* __restrict__ bias, OT* __restrict__ out,
                    float alpha) {
  constexpr int BM = 128, BN = 64, BK = 32;
  constexpr int ASTR = 40;  // padded row stride (halves) -> conflict-free b128
  constexpr int BSTR = 40;
  constexpr int N = D_MODEL, K = D_MODEL;

  __shared__ __bf16 As[BM * ASTR];  // A tile, row-major [128][40]
  __shared__ __bf16 Bt[BN * BSTR];  // W tile transposed: Bt[n][k], [64][40]

  const int t = threadIdx.x;
  const int lane = t & 31, w = t >> 5;
  const int lg = lane >> 4, lr = lane & 15;  // lane group / lane-in-group
  const int gRow0 = blockIdx.y * BM, gCol0 = blockIdx.x * BN;
  const int wm = (w & 3) << 5;   // wave M offset (4 waves * 32 rows)
  const int wn = (w >> 2) << 5;  // wave N offset (2 waves * 32 cols)

  f32x8 acc[2][2] = {};

  for (int kt = 0; kt < K; kt += BK) {
    __syncthreads();
    // --- stage A tile (convert to bf16): thread -> row t/2, 16 halves ------
    {
      int row = t >> 1, c0 = (t & 1) << 4;
      const AT* ap = A + (size_t)(gRow0 + row) * K + kt + c0;
#pragma unroll
      for (int i = 0; i < 16; ++i)
        As[row * ASTR + c0 + i] = (__bf16)(float)ap[i];
    }
    // --- stage W tile transposed: Bt[n][k] = W[kt+k][gCol0+n] --------------
    {
      int k = t >> 3, n0 = (t & 7) << 3;
      const float* wp = W + (size_t)(kt + k) * N + gCol0 + n0;
#pragma unroll
      for (int i = 0; i < 8; ++i)
        Bt[(n0 + i) * BSTR + k] = (__bf16)wp[i];
    }
    if (kt + BK < K) {  // prefetch next tiles -> global_prefetch_b8
      __builtin_prefetch((const void*)(A + (size_t)(gRow0 + (t >> 1)) * K + kt + BK), 0, 1);
      __builtin_prefetch((const void*)(W + (size_t)(kt + BK + (t >> 3)) * N + gCol0 + ((t & 7) << 3)), 0, 1);
    }
    __syncthreads();

    // --- fragment loads (documented 16-bit A/B VGPR layouts) ---------------
    Frag16 af[2], bfr[2];
#pragma unroll
    for (int im = 0; im < 2; ++im) {
      // A: lane<16 row=m, K in {0..7,16..23}; lane>=16 K in {8..15,24..31}
      int base = (wm + (im << 4) + lr) * ASTR + (lg << 3);
      af[im].h[0] = *(const bf16x8*)&As[base];
      af[im].h[1] = *(const bf16x8*)&As[base + 16];
    }
#pragma unroll
    for (int in_ = 0; in_ < 2; ++in_) {
      // B: lane<16 col=n, K in {0..7,16..23}; lane>=16 K in {8..15,24..31}
      int base = (wn + (in_ << 4) + lr) * BSTR + (lg << 3);
      bfr[in_].h[0] = *(const bf16x8*)&Bt[base];
      bfr[in_].h[1] = *(const bf16x8*)&Bt[base + 16];
    }
#pragma unroll
    for (int im = 0; im < 2; ++im)
#pragma unroll
      for (int in_ = 0; in_ < 2; ++in_)
        acc[im][in_] = wmma_bf16(af[im].v, bfr[in_].v, acc[im][in_]);
  }

  // --- epilogue: C-frag layout: lane<16 col=lane rows r; lane>=16 rows r+8 -
#pragma unroll
  for (int im = 0; im < 2; ++im)
#pragma unroll
    for (int in_ = 0; in_ < 2; ++in_) {
      int col = gCol0 + wn + (in_ << 4) + lr;
      float bb = bias[col];
#pragma unroll
      for (int r = 0; r < 8; ++r) {
        int row = gRow0 + wm + (im << 4) + r + (lg << 3);
        out[(size_t)row * N + col] = (OT)(alpha * acc[im][in_][r] + bb);
      }
    }
}

// ---------------------------------------------------------------------------
// Flash attention over bf16 Q/K/V stored [B, S, D], heads interleaved
// (head h = columns h*64 .. h*64+63, contiguous per row). Q pre-scaled by
// 1/sqrt(dk). Block = 128 threads = 4 waves; each wave owns 16 query rows;
// loop over 64-key blocks with online softmax:
//   - row max: ds_swizzle SWAPX1/2/4/8 butterflies (4 batched waits)
//   - row sum: P x ones via WMMA (no lane shuffles)
//   - P re-layout C-frag -> A-frag through per-wave LDS scratch
// ---------------------------------------------------------------------------
__global__ __launch_bounds__(128)
void flash_attn_bf16(const __bf16* __restrict__ Qb, const __bf16* __restrict__ Kb,
                     const __bf16* __restrict__ Vb, __bf16* __restrict__ Ctx) {
  constexpr int KSTR = 72;  // K tile [64 keys][64 dk] padded
  constexpr int VSTR = 72;  // Vt tile [64 dk][64 keys] padded
  constexpr int PSTR = 72;  // per-wave P scratch [16 rows][64 keys] padded

  __shared__ __bf16 Ks[64 * KSTR];
  __shared__ __bf16 Vt[64 * VSTR];
  __shared__ __bf16 Ps[4 * 16 * PSTR];

  const int t = threadIdx.x, lane = t & 31, w = t >> 5;
  const int lg = lane >> 4, lr = lane & 15;
  const int b = blockIdx.z, h = blockIdx.y;
  const size_t bS = (size_t)b * SEQ;
  const int hOff = h * DK;
  const int qRow = blockIdx.x * 64 + w * 16;

  // Q fragments (A layout), loaded straight from global, held in VGPRs.
  Frag16 qf[2];
  {
    const __bf16* qp = Qb + (bS + qRow + lr) * D_MODEL + hOff + (lg << 3);
    qf[0].h[0] = *(const bf16x8*)(qp);
    qf[0].h[1] = *(const bf16x8*)(qp + 16);
    qf[1].h[0] = *(const bf16x8*)(qp + 32);
    qf[1].h[1] = *(const bf16x8*)(qp + 48);
  }

  Frag16 ones;  // all-ones B matrix for row-sum WMMA
#pragma unroll
  for (int i = 0; i < 16; ++i) ones.v[i] = (__bf16)1.0f;

  f32x8 accO[4] = {};
  float mrow[8], lrow[8];
#pragma unroll
  for (int r = 0; r < 8; ++r) { mrow[r] = -1e30f; lrow[r] = 0.0f; }

  __bf16* PsW = &Ps[w * 16 * PSTR];

  for (int kb = 0; kb < SEQ; kb += 64) {
    __syncthreads();
    // --- stage K block [64][64]: pure 128-bit copies -----------------------
    {
      int row = t >> 1, c0 = (t & 1) << 5;  // 2 threads/row, 32 halves each
      const __bf16* gp = Kb + (bS + kb + row) * D_MODEL + hOff + c0;
#if HAVE_ASYNC_LDS
#pragma unroll
      for (int j = 0; j < 4; ++j)
        __builtin_amdgcn_global_load_async_to_lds_b128(
            (as1_v4i_p)(gp + 8 * j),
            (as3_v4i_p)&Ks[row * KSTR + c0 + 8 * j], 0, 0);
#else
#pragma unroll
      for (int j = 0; j < 4; ++j)
        *(bf16x8*)&Ks[row * KSTR + c0 + 8 * j] = *(const bf16x8*)(gp + 8 * j);
#endif
    }
    // --- stage V block transposed: Vt[d][k] = V[kb+k][h*64+d] --------------
    {
      int k = t >> 1, d0 = (t & 1) << 5;
      const __bf16* vp = Vb + (bS + kb + k) * D_MODEL + hOff + d0;
#pragma unroll
      for (int i = 0; i < 32; ++i) Vt[(d0 + i) * VSTR + k] = vp[i];
    }
#if HAVE_ASYNC_LDS
    asm volatile("s_wait_asynccnt 0x0" ::: "memory");
#endif
    __syncthreads();

    // --- scores: 4 tiles of 16 keys, dk=64 -> 2 WMMAs each -----------------
    f32x8 s[4];
#pragma unroll
    for (int g = 0; g < 4; ++g) {
      Frag16 kf;
      int keyr = (g << 4) + lr;
      kf.h[0] = *(const bf16x8*)&Ks[keyr * KSTR + (lg << 3)];
      kf.h[1] = *(const bf16x8*)&Ks[keyr * KSTR + (lg << 3) + 16];
      f32x8 z = {};
      z = wmma_bf16(qf[0].v, kf.v, z);
      kf.h[0] = *(const bf16x8*)&Ks[keyr * KSTR + 32 + (lg << 3)];
      kf.h[1] = *(const bf16x8*)&Ks[keyr * KSTR + 32 + (lg << 3) + 16];
      s[g] = wmma_bf16(qf[1].v, kf.v, z);
    }

    // --- row max: local max of 4 tiles, then 4 batched swizzle steps -------
    float mx[8];
#pragma unroll
    for (int r = 0; r < 8; ++r)
      mx[r] = fmaxf(fmaxf(s[0][r], s[1][r]), fmaxf(s[2][r], s[3][r]));
    swz_max8<0x041f>(mx);  // SWAPX1
    swz_max8<0x081f>(mx);  // SWAPX2
    swz_max8<0x101f>(mx);  // SWAPX4
    swz_max8<0x201f>(mx);  // SWAPX8

    float corr[8];
#pragma unroll
    for (int r = 0; r < 8; ++r) {
      float mnew = fmaxf(mrow[r], mx[r]);
      corr[r] = __expf(mrow[r] - mnew);
      mrow[r] = mnew;
    }

    // --- exp + spill P (C-frag layout) to per-wave LDS scratch -------------
#pragma unroll
    for (int g = 0; g < 4; ++g)
#pragma unroll
      for (int r = 0; r < 8; ++r) {
        float p = __expf(s[g][r] - mrow[r]);
        PsW[(r + (lg << 3)) * PSTR + (g << 4) + lr] = (__bf16)p;
      }
    asm volatile("s_wait_dscnt 0x0" ::: "memory");

    // --- reload P in A-frag layout: 2 frags (keys 0-31, 32-63) -------------
    Frag16 pf[2];
#pragma unroll
    for (int q = 0; q < 2; ++q) {
      int base = lr * PSTR + (q << 5) + (lg << 3);
      pf[q].h[0] = *(const bf16x8*)&PsW[base];
      pf[q].h[1] = *(const bf16x8*)&PsW[base + 16];
    }

    // --- rescale running O, accumulate P.V (8 WMMAs) -----------------------
#pragma unroll
    for (int c = 0; c < 4; ++c)
#pragma unroll
      for (int r = 0; r < 8; ++r) accO[c][r] *= corr[r];
#pragma unroll
    for (int c = 0; c < 4; ++c)
#pragma unroll
      for (int q = 0; q < 2; ++q) {
        Frag16 vf;  // B-frag: lane col = dk chunk c*16+lr, keys {8lg..+7, 16+8lg..+7}
        int base = ((c << 4) + lr) * VSTR + (q << 5) + (lg << 3);
        vf.h[0] = *(const bf16x8*)&Vt[base];
        vf.h[1] = *(const bf16x8*)&Vt[base + 16];
        accO[c] = wmma_bf16(pf[q].v, vf.v, accO[c]);
      }

    // --- row sums via WMMA against ones (broadcast to every lane) ----------
    f32x8 rs = {};
    rs = wmma_bf16(pf[0].v, ones.v, rs);
    rs = wmma_bf16(pf[1].v, ones.v, rs);
#pragma unroll
    for (int r = 0; r < 8; ++r) lrow[r] = lrow[r] * corr[r] + rs[r];
  }

  // --- normalize and store ctx (bf16) --------------------------------------
  float inv[8];
#pragma unroll
  for (int r = 0; r < 8; ++r) inv[r] = 1.0f / lrow[r];
#pragma unroll
  for (int c = 0; c < 4; ++c)
#pragma unroll
    for (int r = 0; r < 8; ++r) {
      size_t off = (bS + qRow + r + (lg << 3)) * D_MODEL + hOff + (c << 4) + lr;
      Ctx[off] = (__bf16)(accO[c][r] * inv[r]);
    }
}

// ---------------------------------------------------------------------------
// Launch
// ---------------------------------------------------------------------------
extern "C" void kernel_launch(void* const* d_in, const int* in_sizes, int n_in,
                              void* d_out, int out_size, void* d_ws, size_t ws_size,
                              hipStream_t stream) {
  const float* query  = (const float*)d_in[0];
  const float* key_in = (const float*)d_in[1];
  const float* value  = (const float*)d_in[2];
  const float* Wq = (const float*)d_in[3];
  const float* bq = (const float*)d_in[4];
  const float* Wk = (const float*)d_in[5];
  const float* bk = (const float*)d_in[6];
  const float* Wv = (const float*)d_in[7];
  const float* bv = (const float*)d_in[8];
  const float* Wo = (const float*)d_in[9];
  const float* bo = (const float*)d_in[10];
  float* out = (float*)d_out;

  const size_t tok   = (size_t)NB * SEQ;  // 4096 rows
  const size_t elems = tok * D_MODEL;     // 4 Mi elements per tensor
  __bf16* Qb  = (__bf16*)d_ws;
  __bf16* Kb  = Qb + elems;
  __bf16* Vb  = Kb + elems;
  __bf16* Ctx = Vb + elems;               // total 32 MB of ws

  dim3 gGrid(D_MODEL / 64, (unsigned)(tok / 128));  // (16, 32)
  dim3 gBlk(256);
  // Q pre-scaled by 1/sqrt(dk) = 0.125 so attention needs no extra multiply.
  gemm_bf16_wmma<float, __bf16><<<gGrid, gBlk, 0, stream>>>(query,  Wq, bq, Qb, 0.125f);
  gemm_bf16_wmma<float, __bf16><<<gGrid, gBlk, 0, stream>>>(key_in, Wk, bk, Kb, 1.0f);
  gemm_bf16_wmma<float, __bf16><<<gGrid, gBlk, 0, stream>>>(value,  Wv, bv, Vb, 1.0f);

  dim3 fGrid(SEQ / 64, NH, NB);  // (32, 16, 2)
  flash_attn_bf16<<<fGrid, dim3(128), 0, stream>>>(Qb, Kb, Vb, Ctx);

  gemm_bf16_wmma<__bf16, float><<<gGrid, gBlk, 0, stream>>>(Ctx, Wo, bo, out, 1.0f);
}
